// GCN_77275051590253
// MI455X (gfx1250) — compile-verified, matrix operand order checked by reading, not run on hardware
//
#include <hip/hip_runtime.h>

typedef __bf16 v16bf __attribute__((ext_vector_type(16)));
typedef __bf16 v2bf  __attribute__((ext_vector_type(2)));
typedef float  v8f   __attribute__((ext_vector_type(8)));

union BFVec { v2bf p[8]; v16bf b; };

// Packed fp32 -> bf16 (RNE) conversion: use the hardware packed cvt when the
// builtin exists, otherwise plain casts (backend still picks hw cvt if legal).
__device__ __forceinline__ v2bf pk2(float lo, float hi) {
#if __has_builtin(__builtin_amdgcn_cvt_pk_bf16_f32)
    auto t = __builtin_amdgcn_cvt_pk_bf16_f32(lo, hi);
    return __builtin_bit_cast(v2bf, t);
#else
    v2bf r;
    r[0] = (__bf16)lo;
    r[1] = (__bf16)hi;
    return r;
#endif
}

__device__ __forceinline__ unsigned short f2bf(float f) {
    union { float f; unsigned int u; } a; a.f = f;
    unsigned int r = a.u + 0x7fffu + ((a.u >> 16) & 1u);  // round-to-nearest-even
    return (unsigned short)(r >> 16);
}

// ---------------- simple utility kernels ----------------

__global__ void k_zero(float* p, int n) {
    int i = blockIdx.x * blockDim.x + threadIdx.x;
    if (i < n) p[i] = 0.0f;
}

__global__ void k_deg(const int* __restrict__ dst, float* __restrict__ deg, int E) {
    int i = blockIdx.x * blockDim.x + threadIdx.x;
    if (i < E) atomicAdd(&deg[dst[i]], 1.0f);
}

__global__ void k_dinv(const float* __restrict__ deg, float* __restrict__ dinv, int n) {
    int i = blockIdx.x * blockDim.x + threadIdx.x;
    if (i < n) dinv[i] = rsqrtf(deg[i] + 1.0f);   // +1 for self-loop
}

// Pre-swizzle W1 (512x32) and W2 (32x16) into the per-lane WMMA B-fragment
// layout: [tile][kb][lane][16] bf16, where element j of lane L covers
// K = kb*32 + (j>>3)*16 + (L<16?0:8) + (j&7), N = tile*16 + (L&15).
__global__ void k_prepB(const float* __restrict__ W1, const float* __restrict__ W2,
                        unsigned short* __restrict__ bf1, unsigned short* __restrict__ bf2) {
    for (int idx = threadIdx.x; idx < 2 * 16 * 32 * 16; idx += blockDim.x) {
        int tile = idx / 8192;
        int rem  = idx % 8192;
        int kb   = rem / 512;
        int r2   = rem % 512;
        int lane = r2 / 16;
        int j    = r2 % 16;
        int col  = tile * 16 + (lane & 15);
        int sel  = (lane < 16) ? 0 : 8;
        int k    = kb * 32 + (j >> 3) * 16 + sel + (j & 7);
        bf1[idx] = f2bf(W1[k * 32 + col]);
    }
    for (int idx = threadIdx.x; idx < 512; idx += blockDim.x) {
        int lane = idx / 16;
        int j    = idx % 16;
        int col  = lane & 15;
        int sel  = (lane < 16) ? 0 : 8;
        int k    = (j >> 3) * 16 + sel + (j & 7);
        bf2[idx] = f2bf(W2[k * 16 + col]);
    }
}

// Build a 16-element bf16 A-fragment from two 8-float groups (4x float4).
__device__ __forceinline__ v16bf make_afrag(float4 f0, float4 f1, float4 f2, float4 f3) {
    BFVec a;
    a.p[0] = pk2(f0.x, f0.y); a.p[1] = pk2(f0.z, f0.w);
    a.p[2] = pk2(f1.x, f1.y); a.p[3] = pk2(f1.z, f1.w);
    a.p[4] = pk2(f2.x, f2.y); a.p[5] = pk2(f2.z, f2.w);
    a.p[6] = pk2(f3.x, f3.y); a.p[7] = pk2(f3.z, f3.w);
    return a.b;
}

// ---------------- GEMM1: h1s = (x @ W1) * dinv[row] ----------------
// One wave per 16-row tile, full 32-col output via two accumulators.
__global__ void k_gemm1(const float* __restrict__ x, const unsigned short* __restrict__ bfrag,
                        const float* __restrict__ dinv, float* __restrict__ h1s,
                        int N, int ntiles) {
    int wave = (blockIdx.x * blockDim.x + threadIdx.x) >> 5;
    int lane = threadIdx.x & 31;
    if (wave >= ntiles) return;                 // wave-uniform: EXEC all-ones for WMMA
    int sel  = (lane >> 4) << 3;                // 0 or 8
    int row  = wave * 16 + (lane & 15);
    int rowc = row < N ? row : N - 1;
    const float* xr = x + rowc * 512;

    v8f c0 = {};
    v8f c1 = {};
    const v16bf* b0p = (const v16bf*)(bfrag);                  // cols 0..15
    const v16bf* b1p = (const v16bf*)(bfrag + 16 * 32 * 16);   // cols 16..31

    #pragma unroll
    for (int kb = 0; kb < 16; ++kb) {
        const float* g0 = xr + kb * 32 + sel;
        const float* g1 = xr + kb * 32 + 16 + sel;
        float4 f0 = *(const float4*)(g0);
        float4 f1 = *(const float4*)(g0 + 4);
        float4 f2 = *(const float4*)(g1);
        float4 f3 = *(const float4*)(g1 + 4);
        v16bf a = make_afrag(f0, f1, f2, f3);

        v16bf bb0 = b0p[kb * 32 + lane];
        v16bf bb1 = b1p[kb * 32 + lane];
        c0 = __builtin_amdgcn_wmma_f32_16x16x32_bf16(false, a, false, bb0, (short)0, c0, false, false);
        c1 = __builtin_amdgcn_wmma_f32_16x16x32_bf16(false, a, false, bb1, (short)0, c1, false, false);
    }

    int rbase = wave * 16 + sel;      // C/D: VGPR r -> row rbase + r
    int col   = lane & 15;
    #pragma unroll
    for (int r = 0; r < 8; ++r) {
        int m = rbase + r;
        if (m < N) {
            float dv = dinv[m];
            h1s[m * 32 + col]      = c0[r] * dv;
            h1s[m * 32 + 16 + col] = c1[r] * dv;
        }
    }
}

// ---------------- aggregation layer 1 ----------------

__global__ void k_init1(const float* __restrict__ h1s, const float* __restrict__ dinv,
                        float* __restrict__ out1, int n) {   // self-loop: h1s * dinv (h*dinv^2)
    int i = blockIdx.x * blockDim.x + threadIdx.x;
    if (i < n) out1[i] = h1s[i] * dinv[i >> 5];
}

__global__ void k_edge1(const int* __restrict__ src, const int* __restrict__ dst,
                        const float* __restrict__ h1s, const float* __restrict__ dinv,
                        float* __restrict__ out1, int E) {
    int t = blockIdx.x * blockDim.x + threadIdx.x;
    int e = t >> 3;
    if (e >= E) return;
    int c = (t & 7) << 2;
    int s = src[e], d = dst[e];
    float w = dinv[d];
    float4 v = *(const float4*)(h1s + s * 32 + c);
    float* o = out1 + d * 32 + c;
    atomicAdd(o + 0, v.x * w);
    atomicAdd(o + 1, v.y * w);
    atomicAdd(o + 2, v.z * w);
    atomicAdd(o + 3, v.w * w);
}

__global__ void k_bias_relu(float* __restrict__ out1, const float* __restrict__ b1, int n) {
    int i = blockIdx.x * blockDim.x + threadIdx.x;
    if (i < n) {
        float v = out1[i] + b1[i & 31];
        out1[i] = v > 0.0f ? v : 0.0f;
    }
}

// ---------------- GEMM2: h2s = (relu_out1 @ W2) * dinv[row] (K=32, one WMMA) ----------------
__global__ void k_gemm2(const float* __restrict__ h, const unsigned short* __restrict__ bfrag,
                        const float* __restrict__ dinv, float* __restrict__ h2s,
                        int N, int ntiles) {
    int wave = (blockIdx.x * blockDim.x + threadIdx.x) >> 5;
    int lane = threadIdx.x & 31;
    if (wave >= ntiles) return;
    int sel  = (lane >> 4) << 3;
    int row  = wave * 16 + (lane & 15);
    int rowc = row < N ? row : N - 1;
    const float* hr = h + rowc * 32;

    float4 f0 = *(const float4*)(hr + sel);
    float4 f1 = *(const float4*)(hr + sel + 4);
    float4 f2 = *(const float4*)(hr + 16 + sel);
    float4 f3 = *(const float4*)(hr + 16 + sel + 4);
    v16bf a = make_afrag(f0, f1, f2, f3);

    v16bf bb = ((const v16bf*)bfrag)[lane];
    v8f c = {};
    c = __builtin_amdgcn_wmma_f32_16x16x32_bf16(false, a, false, bb, (short)0, c, false, false);

    int rbase = wave * 16 + sel;
    int col   = lane & 15;
    #pragma unroll
    for (int r = 0; r < 8; ++r) {
        int m = rbase + r;
        if (m < N) h2s[m * 16 + col] = c[r] * dinv[m];
    }
}

// ---------------- aggregation layer 2 (accumulate directly into d_out) ----------------

__global__ void k_init2(const float* __restrict__ h2s, const float* __restrict__ dinv,
                        float* __restrict__ out, int n) {
    int i = blockIdx.x * blockDim.x + threadIdx.x;
    if (i < n) out[i] = h2s[i] * dinv[i >> 4];
}

__global__ void k_edge2(const int* __restrict__ src, const int* __restrict__ dst,
                        const float* __restrict__ h2s, const float* __restrict__ dinv,
                        float* __restrict__ out, int E) {
    int t = blockIdx.x * blockDim.x + threadIdx.x;
    int e = t >> 2;
    if (e >= E) return;
    int c = (t & 3) << 2;
    int s = src[e], d = dst[e];
    float w = dinv[d];
    float4 v = *(const float4*)(h2s + s * 16 + c);
    float* o = out + d * 16 + c;
    atomicAdd(o + 0, v.x * w);
    atomicAdd(o + 1, v.y * w);
    atomicAdd(o + 2, v.z * w);
    atomicAdd(o + 3, v.w * w);
}

__global__ void k_lsm(float* __restrict__ out, const float* __restrict__ b2, int N) {
    int i = blockIdx.x * blockDim.x + threadIdx.x;
    if (i >= N) return;
    float v[16];
    float m = -1e30f;
    #pragma unroll
    for (int f = 0; f < 16; ++f) {
        v[f] = out[i * 16 + f] + b2[f];
        m = fmaxf(m, v[f]);
    }
    float s = 0.0f;
    #pragma unroll
    for (int f = 0; f < 16; ++f) s += expf(v[f] - m);
    float l = m + logf(s);
    #pragma unroll
    for (int f = 0; f < 16; ++f) out[i * 16 + f] = v[f] - l;
}

// ---------------- launch ----------------

extern "C" void kernel_launch(void* const* d_in, const int* in_sizes, int n_in,
                              void* d_out, int out_size, void* d_ws, size_t ws_size,
                              hipStream_t stream) {
    const float* x  = (const float*)d_in[0];
    const int*   ei = (const int*)d_in[1];
    const float* W1 = (const float*)d_in[2];
    const float* b1 = (const float*)d_in[3];
    const float* W2 = (const float*)d_in[4];
    const float* b2 = (const float*)d_in[5];
    float* out = (float*)d_out;

    const int N = in_sizes[0] / 512;   // 100000
    const int E = in_sizes[1] / 2;     // 1600000
    const int* src = ei;
    const int* dst = ei + E;

    // workspace layout (floats padded to 64-elem boundaries)
    const int NP = (N + 63) & ~63;
    float* deg  = (float*)d_ws;
    float* dinv = deg + NP;
    float* h1s  = dinv + NP;                       // N*32
    float* out1 = h1s + N * 32;                    // N*32
    float* h2s  = out1 + N * 32;                   // N*16
    unsigned short* bf1 = (unsigned short*)(h2s + N * 16);   // 2*16*32*16 = 16384
    unsigned short* bf2 = bf1 + 16384;                       // 512

    const int TB = 256;
    const int ntiles = (N + 15) / 16;
    dim3 blk(TB);

    k_zero<<<dim3((N + TB - 1) / TB), blk, 0, stream>>>(deg, N);
    k_prepB<<<dim3(1), blk, 0, stream>>>(W1, W2, bf1, bf2);
    k_deg<<<dim3((E + TB - 1) / TB), blk, 0, stream>>>(dst, deg, E);
    k_dinv<<<dim3((N + TB - 1) / TB), blk, 0, stream>>>(deg, dinv, N);

    k_gemm1<<<dim3((ntiles * 32 + TB - 1) / TB), blk, 0, stream>>>(x, bf1, dinv, h1s, N, ntiles);
    k_init1<<<dim3((N * 32 + TB - 1) / TB), blk, 0, stream>>>(h1s, dinv, out1, N * 32);
    k_edge1<<<dim3((E * 8 + TB - 1) / TB), blk, 0, stream>>>(src, dst, h1s, dinv, out1, E);
    k_bias_relu<<<dim3((N * 32 + TB - 1) / TB), blk, 0, stream>>>(out1, b1, N * 32);

    k_gemm2<<<dim3((ntiles * 32 + TB - 1) / TB), blk, 0, stream>>>(out1, bf2, dinv, h2s, N, ntiles);
    k_init2<<<dim3((N * 16 + TB - 1) / TB), blk, 0, stream>>>(h2s, dinv, out, N * 16);
    k_edge2<<<dim3((E * 4 + TB - 1) / TB), blk, 0, stream>>>(src, dst, h2s, dinv, out, E);
    k_lsm<<<dim3((N + TB - 1) / TB), blk, 0, stream>>>(out, b2, N);
}
